// GraphMambaModel_37726992728373
// MI455X (gfx1250) — matrix operand order repaired and checked by reading, not compiled, not run on hardware
//
#include <hip/hip_runtime.h>
#include <hip/hip_bf16.h>

// ---------------------------------------------------------------------------
// GraphMamba forward for MI455X (gfx1250), wave32, WMMA bf16 path.
// Weights pre-packed to bf16 fragment-major; guard-free GEMM inner loops.
// GEMM: 32x64 output per wave = 2 M-tiles x 4 N-tiles = 8 WMMAs / k-step.
// ---------------------------------------------------------------------------

#define NN   16384
#define EE   262144
#define FF   32
#define HH   128
#define BB   16
#define DI   256
#define DS   16
#define DC   4
#define DTR  8
#define TOTC 1552   // 1+4+3+512+512+512+8
#define EPSV 1e-5f
#define LDBL 64     // padded row stride of dbl

typedef __attribute__((ext_vector_type(16))) __bf16 v16bf;
typedef __attribute__((ext_vector_type(8)))  float  v8f;

// ---------------------------------------------------------------------------
// Weight pack: f32 (Korig x Norig, ld=Norig) -> bf16 fragment-major with
// zero padding to (Kpad x Npad).  Element for K=k, col=c stored at
//   [ ((k/32)*Npad + c)*32 + (k%32) ]
// so a lane's 16-element B fragment (col, khalf) is one contiguous 32B chunk.
// ---------------------------------------------------------------------------
__global__ void pack_w(const float* __restrict__ W, int Korig, int Norig,
                       int Kpad, int Npad, __bf16* __restrict__ out) {
    int i = blockIdx.x * blockDim.x + threadIdx.x;      // over Kpad*Npad
    if (i >= Kpad * Npad) return;
    int k = i / Npad, c = i % Npad;
    float v = (k < Korig && c < Norig) ? W[(long)k * Norig + c] : 0.f;
    out[(((long)(k >> 5) * Npad + c) << 5) + (k & 31)] = (__bf16)v;
}

// ---------------------------------------------------------------------------
// WMMA GEMM, 32x64 output tile per wave (2 M x 4 N register-blocked).
// Caller guarantees: M%32==0, K%32==0, Nn%64==0, lda multiple of 8.
// C = act(A[M,K] @ W[K,Nn] + bias), bf16 inputs / f32 accumulate.
// Fragment k order: contiguous, k = 16*half + s, same for A and B (wave32).
// act: 0=none, 1=relu, 2=softplus
// ---------------------------------------------------------------------------
__device__ __forceinline__ v16bf load_a_frag(const float* __restrict__ Arow, int k0) {
    float4 p0 = *(const float4*)(Arow + k0);
    float4 p1 = *(const float4*)(Arow + k0 + 4);
    float4 p2 = *(const float4*)(Arow + k0 + 8);
    float4 p3 = *(const float4*)(Arow + k0 + 12);
    v16bf a;
    a[0]  = (__bf16)p0.x; a[1]  = (__bf16)p0.y;
    a[2]  = (__bf16)p0.z; a[3]  = (__bf16)p0.w;
    a[4]  = (__bf16)p1.x; a[5]  = (__bf16)p1.y;
    a[6]  = (__bf16)p1.z; a[7]  = (__bf16)p1.w;
    a[8]  = (__bf16)p2.x; a[9]  = (__bf16)p2.y;
    a[10] = (__bf16)p2.z; a[11] = (__bf16)p2.w;
    a[12] = (__bf16)p3.x; a[13] = (__bf16)p3.y;
    a[14] = (__bf16)p3.z; a[15] = (__bf16)p3.w;
    return a;
}

__global__ __launch_bounds__(32)
void wmma_gemm8(const float* __restrict__ A, int lda,
                const __bf16* __restrict__ Wp, int Nn,
                const float* __restrict__ bias,
                float* __restrict__ C, int ldc,
                int K, int act)
{
    const int lane = threadIdx.x;             // one wave per block
    const int row0 = blockIdx.y * 32;
    const int col0 = blockIdx.x * 64;
    const int half = lane >> 4;
    const int l15  = lane & 15;

    // A row pointers pre-offset by 16*half: fragment = one run of 16 floats
    const float* __restrict__ Arow0 = A + (long)(row0 + l15) * lda + 16 * half;
    const float* __restrict__ Arow1 = Arow0 + (long)16 * lda;

    v8f acc0[4] = {v8f{}, v8f{}, v8f{}, v8f{}};
    v8f acc1[4] = {v8f{}, v8f{}, v8f{}, v8f{}};

    for (int k0 = 0; k0 < K; k0 += 32) {
        v16bf a0 = load_a_frag(Arow0, k0);
        v16bf a1 = load_a_frag(Arow1, k0);

        const __bf16* __restrict__ Wk = Wp + (((long)(k0 >> 5) * Nn) << 5)
                                           + (half << 4);
#pragma unroll
        for (int t = 0; t < 4; ++t) {
            int col = col0 + t * 16 + l15;
            v16bf b = *(const v16bf*)(Wk + ((long)col << 5));
            acc0[t] = __builtin_amdgcn_wmma_f32_16x16x32_bf16(
                false, a0, false, b, (short)0, acc0[t], false, false);
            acc1[t] = __builtin_amdgcn_wmma_f32_16x16x32_bf16(
                false, a1, false, b, (short)0, acc1[t], false, false);
        }
    }

    // ---- store 8 tiles ----
#pragma unroll
    for (int t = 0; t < 4; ++t) {
        int col = col0 + t * 16 + l15;
        float bv = bias ? bias[col] : 0.f;
#pragma unroll
        for (int r = 0; r < 8; ++r) {
            int row = row0 + r + 8 * half;
            float v = acc0[t][r] + bv;
            if (act == 1)      v = fmaxf(v, 0.f);
            else if (act == 2) v = (v > 20.f) ? v : log1pf(expf(v));
            C[(long)row * ldc + col] = v;

            float v2 = acc1[t][r] + bv;
            if (act == 1)      v2 = fmaxf(v2, 0.f);
            else if (act == 2) v2 = (v2 > 20.f) ? v2 : log1pf(expf(v2));
            C[(long)(row + 16) * ldc + col] = v2;
        }
    }
}

// ---------------------------------------------------------------------------
// Small helper kernels
// ---------------------------------------------------------------------------
__global__ void fill_kernel(float* p, float v, int n) {
    int i = blockIdx.x * blockDim.x + threadIdx.x;
    if (i < n) p[i] = v;
}

__global__ void deg_accum(const int* __restrict__ dst, float* deg, int e) {
    int i = blockIdx.x * blockDim.x + threadIdx.x;
    if (i < e) atomicAdd(&deg[dst[i]], 1.0f);
}

__global__ void rsqrt_kernel(float* p, int n) {
    int i = blockIdx.x * blockDim.x + threadIdx.x;
    if (i < n) p[i] = rsqrtf(p[i]);        // deg >= 1 (self-loop)
}

// out[n,:] = xw[n,:]*dinv[n]^2   (self-loop contribution)
__global__ void gcn_self(const float* __restrict__ xw,
                         const float* __restrict__ dinv,
                         float* __restrict__ out, int total) {
    int i = blockIdx.x * blockDim.x + threadIdx.x;
    if (i < total) {
        float d = dinv[i / HH];
        out[i] = xw[i] * d * d;
    }
}

// scatter-add edge messages; thread = (edge, 4-feature group)
__global__ void gcn_edges(const int* __restrict__ src, const int* __restrict__ dst,
                          const float* __restrict__ xw, const float* __restrict__ dinv,
                          float* __restrict__ out, int e) {
    int t = blockIdx.x * blockDim.x + threadIdx.x;
    int eidx = t >> 5;             // H/4 = 32 groups per edge
    if (eidx >= e) return;
    int fg = (t & 31) * 4;
    int s = src[eidx], d = dst[eidx];
    float c = dinv[s] * dinv[d];
    const float* xr = xw + (long)s * HH + fg;
    float* orow = out + (long)d * HH + fg;
#pragma unroll
    for (int j = 0; j < 4; ++j) atomicAdd(&orow[j], xr[j] * c);
}

__global__ void bias_relu(float* __restrict__ p, const float* __restrict__ b, int total) {
    int i = blockIdx.x * blockDim.x + threadIdx.x;
    if (i < total) p[i] = fmaxf(p[i] + b[i % HH], 0.f);
}

// causal depthwise conv (k=4) over node axis + silu.  x read from xz[:, :DI] (ld 512)
__global__ void conv_silu(const float* __restrict__ xz,
                          const float* __restrict__ w,   // DI x DC
                          const float* __restrict__ b,   // DI
                          float* __restrict__ xo) {
    int i = blockIdx.x * blockDim.x + threadIdx.x;       // over N*DI
    if (i >= NN * DI) return;
    int n = i / DI, c = i % DI;
    float acc = b[c];
#pragma unroll
    for (int k = 0; k < DC; ++k) {
        int t = n + k - (DC - 1);
        float xv = (t >= 0) ? xz[(long)t * (2 * DI) + c] : 0.f;
        acc += xv * w[c * DC + k];
    }
    xo[i] = acc / (1.f + expf(-acc));                    // silu
}

// Selective scan: thread c owns channel c, 16 states in registers, sequential in t.
__global__ __launch_bounds__(DI)
void mamba_scan(const float* __restrict__ dt,     // N x DI
                const float* __restrict__ dbl,    // N x LDBL (B at 8, C at 24)
                const float* __restrict__ x,      // N x DI
                const float* __restrict__ xz,     // N x 512, z = cols 256:
                const float* __restrict__ A_log,  // DI x DS
                const float* __restrict__ D_p,    // DI
                float* __restrict__ y)            // N x DI
{
    int c = threadIdx.x;
    float A[DS], hst[DS];
#pragma unroll
    for (int s = 0; s < DS; ++s) { A[s] = -expf(A_log[c * DS + s]); hst[s] = 0.f; }
    float Dp = D_p[c];
    for (int t = 0; t < NN; ++t) {
        float dtv = dt[(long)t * DI + c];
        float xv  = x[(long)t * DI + c];
        float zv  = xz[(long)t * (2 * DI) + DI + c];
        const float* Bm = dbl + (long)t * LDBL + DTR;
        const float* Cm = Bm + DS;
        float yv = 0.f;
#pragma unroll
        for (int s = 0; s < DS; ++s) {
            hst[s] = expf(dtv * A[s]) * hst[s] + dtv * Bm[s] * xv;
            yv += hst[s] * Cm[s];
        }
        float sz = zv / (1.f + expf(-zv));
        y[(long)t * DI + c] = (yv + Dp * xv) * sz;
    }
}

// hh = hg + hm ; LayerNorm over H=128 ; one wave32 per row (4 feats/lane)
__global__ void add_layernorm(const float* __restrict__ hg, const float* __restrict__ hm,
                              const float* __restrict__ gamma, const float* __restrict__ beta,
                              float* __restrict__ out) {
    int row  = (blockIdx.x * blockDim.x + threadIdx.x) >> 5;
    int lane = threadIdx.x & 31;
    if (row >= NN) return;
    const float* a = hg + (long)row * HH;
    const float* b = hm + (long)row * HH;
    float v[4];
    float sum = 0.f;
#pragma unroll
    for (int j = 0; j < 4; ++j) { v[j] = a[lane + 32 * j] + b[lane + 32 * j]; sum += v[j]; }
#pragma unroll
    for (int off = 16; off > 0; off >>= 1) sum += __shfl_xor(sum, off, 32);
    float mu = sum * (1.f / HH);
    float var = 0.f;
#pragma unroll
    for (int j = 0; j < 4; ++j) { float d = v[j] - mu; var += d * d; }
#pragma unroll
    for (int off = 16; off > 0; off >>= 1) var += __shfl_xor(var, off, 32);
    float inv = rsqrtf(var * (1.f / HH) + EPSV);
#pragma unroll
    for (int j = 0; j < 4; ++j) {
        int f = lane + 32 * j;
        out[(long)row * HH + f] = (v[j] - mu) * inv * gamma[f] + beta[f];
    }
}

// segment sums for graph pooling
__global__ void seg_accum(const float* __restrict__ hf, const int* __restrict__ batch,
                          float* __restrict__ gesum, float* __restrict__ counts) {
    int i = blockIdx.x * blockDim.x + threadIdx.x;   // over N*H
    if (i >= NN * HH) return;
    int n = i / HH, f = i % HH;
    int bi = batch[n];
    atomicAdd(&gesum[bi * HH + f], hf[i]);
    if (f == 0) atomicAdd(&counts[bi], 1.0f);
}

__global__ void seg_div(float* __restrict__ gesum, const float* __restrict__ counts) {
    int i = blockIdx.x * blockDim.x + threadIdx.x;   // over B*H
    if (i < BB * HH) gesum[i] /= fmaxf(counts[i / HH], 1.0f);
}

// 7 output heads, thread per (graph, output col)
__global__ void heads_kernel(const float* __restrict__ ge,
    const float* Wc, const float* bc, const float* Wh, const float* bh,
    const float* Wt, const float* bt, const float* Wp1, const float* bp1,
    const float* Wp2, const float* bp2, const float* Wd, const float* bd,
    const float* Ws, const float* bs, float* __restrict__ out) {
    int t = blockIdx.x * blockDim.x + threadIdx.x;
    if (t >= BB * TOTC) return;
    int g = t / TOTC, col = t % TOTC;
    const float* W; const float* bias; int c, nc;
    if      (col < 1)    { W = Wc;  bias = bc;  c = col;        nc = 1;   }
    else if (col < 5)    { W = Wh;  bias = bh;  c = col - 1;    nc = 4;   }
    else if (col < 8)    { W = Wt;  bias = bt;  c = col - 5;    nc = 3;   }
    else if (col < 520)  { W = Wp1; bias = bp1; c = col - 8;    nc = 512; }
    else if (col < 1032) { W = Wp2; bias = bp2; c = col - 520;  nc = 512; }
    else if (col < 1544) { W = Wd;  bias = bd;  c = col - 1032; nc = 512; }
    else                 { W = Ws;  bias = bs;  c = col - 1544; nc = 8;   }
    float acc = bias[c];
    const float* gr = ge + g * HH;
#pragma unroll 4
    for (int k = 0; k < HH; ++k) acc += gr[k] * W[k * nc + c];
    out[(long)g * TOTC + col] = acc;
}

// ---------------------------------------------------------------------------
// Launch
// ---------------------------------------------------------------------------
extern "C" void kernel_launch(void* const* d_in, const int* in_sizes, int n_in,
                              void* d_out, int out_size, void* d_ws, size_t ws_size,
                              hipStream_t stream) {
    const float* NF     = (const float*)d_in[0];
    const int*   EI     = (const int*)  d_in[1];
    const int*   batch  = (const int*)  d_in[2];
    const float* W_in   = (const float*)d_in[3];
    const float* b_in   = (const float*)d_in[4];
    const float* W_g1   = (const float*)d_in[5];
    const float* b_g1   = (const float*)d_in[6];
    const float* W_g2   = (const float*)d_in[7];
    const float* b_g2   = (const float*)d_in[8];
    const float* W_inpj = (const float*)d_in[9];
    const float* conv_w = (const float*)d_in[10];
    const float* conv_b = (const float*)d_in[11];
    const float* W_xpj  = (const float*)d_in[12];
    const float* W_dt   = (const float*)d_in[13];
    const float* b_dt   = (const float*)d_in[14];
    const float* A_log  = (const float*)d_in[15];
    const float* D_p    = (const float*)d_in[16];
    const float* W_out  = (const float*)d_in[17];
    const float* gamma  = (const float*)d_in[18];
    const float* beta   = (const float*)d_in[19];
    float* out = (float*)d_out;
    float* ws  = (float*)d_ws;

    // workspace layout (floats); buffers reused across phases
    const long SZH = (long)NN * HH;           // 2,097,152
    float* h0    = ws + 0;                    // [0, 2M)
    float* xw    = ws + SZH;                  // [2M, 4M)
    float* h1    = ws + 2 * SZH;              // [4M, 6M)
    float* hgnn  = ws + 3 * SZH;              // [6M, 8M)
    float* xz    = ws + 4 * SZH;              // [8M, 16M)  N x 512
    float* xbuf  = ws + 8 * SZH;              // [16M, 20M) N x 256
    float* dbl   = ws + 10 * SZH;             // N x 64
    float* dtb   = dbl + (long)NN * LDBL;     // N x 256
    float* dinv  = dtb + (long)NN * DI;       // N
    float* gesum = dinv + NN;                 // B x H
    float* cnts  = gesum + BB * HH;           // B
    // packed bf16 weights
    __bf16* pk_win   = (__bf16*)(cnts + BB);            //  32x128
    __bf16* pk_wg1   = pk_win   + 32 * HH;              // 128x128
    __bf16* pk_wg2   = pk_wg1   + HH * HH;              // 128x128
    __bf16* pk_winpj = pk_wg2   + HH * HH;              // 128x512
    __bf16* pk_wxp   = pk_winpj + HH * 512;             // 256x64 (padded)
    __bf16* pk_wdt   = pk_wxp   + DI * LDBL;            //  32x256 (padded)
    __bf16* pk_wout  = pk_wdt   + 32 * DI;              // 256x128
    float* ybuf  = ws + 0;                    // reuse h0+xw region, N x 256
    float* hmam  = ws + 2 * SZH;              // reuse h1
    float* hfin  = ws + 4 * SZH;              // reuse xz head

    const int* esrc = EI;
    const int* edst = EI + EE;

    // --- pack weights to bf16 fragment-major (zero-padded where needed) ---
    pack_w<<<(32 * HH + 255) / 256, 256, 0, stream>>>(W_in,   FF, HH,  32, HH,  pk_win);
    pack_w<<<(HH * HH + 255) / 256, 256, 0, stream>>>(W_g1,   HH, HH,  HH, HH,  pk_wg1);
    pack_w<<<(HH * HH + 255) / 256, 256, 0, stream>>>(W_g2,   HH, HH,  HH, HH,  pk_wg2);
    pack_w<<<(HH * 512 + 255) / 256, 256, 0, stream>>>(W_inpj, HH, 512, HH, 512, pk_winpj);
    pack_w<<<(DI * LDBL + 255) / 256, 256, 0, stream>>>(W_xpj, DI, DTR + 2 * DS, DI, LDBL, pk_wxp);
    pack_w<<<(32 * DI + 255) / 256, 256, 0, stream>>>(W_dt,  DTR, DI,  32, DI,  pk_wdt);
    pack_w<<<(DI * HH + 255) / 256, 256, 0, stream>>>(W_out,  DI, HH,  DI, HH,  pk_wout);

    // --- degree -> dinv (shared by both GCN layers) ---
    fill_kernel<<<(NN + 255) / 256, 256, 0, stream>>>(dinv, 1.0f, NN);      // self-loops
    deg_accum<<<(EE + 255) / 256, 256, 0, stream>>>(edst, dinv, EE);
    rsqrt_kernel<<<(NN + 255) / 256, 256, 0, stream>>>(dinv, NN);

    // --- h0 = relu(NF @ W_in + b_in) ---
    wmma_gemm8<<<dim3(HH / 64, NN / 32), 32, 0, stream>>>(NF, FF, pk_win, HH, b_in,
                                                          h0, HH, FF, 1);
    // --- GCN layer 1 ---
    wmma_gemm8<<<dim3(HH / 64, NN / 32), 32, 0, stream>>>(h0, HH, pk_wg1, HH, nullptr,
                                                          xw, HH, HH, 0);
    gcn_self<<<(NN * HH + 255) / 256, 256, 0, stream>>>(xw, dinv, h1, NN * HH);
    gcn_edges<<<(EE * 32 + 255) / 256, 256, 0, stream>>>(esrc, edst, xw, dinv, h1, EE);
    bias_relu<<<(NN * HH + 255) / 256, 256, 0, stream>>>(h1, b_g1, NN * HH);

    // --- GCN layer 2 ---
    wmma_gemm8<<<dim3(HH / 64, NN / 32), 32, 0, stream>>>(h1, HH, pk_wg2, HH, nullptr,
                                                          xw, HH, HH, 0);
    gcn_self<<<(NN * HH + 255) / 256, 256, 0, stream>>>(xw, dinv, hgnn, NN * HH);
    gcn_edges<<<(EE * 32 + 255) / 256, 256, 0, stream>>>(esrc, edst, xw, dinv, hgnn, EE);
    bias_relu<<<(NN * HH + 255) / 256, 256, 0, stream>>>(hgnn, b_g2, NN * HH);

    // --- xz = h_gnn @ W_inproj  (N x 512) ---
    wmma_gemm8<<<dim3(512 / 64, NN / 32), 32, 0, stream>>>(hgnn, HH, pk_winpj, 512, nullptr,
                                                           xz, 512, HH, 0);
    // --- causal conv + silu -> xbuf ---
    conv_silu<<<(NN * DI + 255) / 256, 256, 0, stream>>>(xz, conv_w, conv_b, xbuf);

    // --- dbl = x @ W_xproj_padded  (N x 64, cols 40..63 zero) ---
    wmma_gemm8<<<dim3(1, NN / 32), 32, 0, stream>>>(xbuf, DI, pk_wxp, LDBL, nullptr,
                                                    dbl, LDBL, DI, 0);
    // --- dt = softplus(dbl[:, :32] @ Wdt_pad + b_dt)  (pad rows are zero) ---
    wmma_gemm8<<<dim3(DI / 64, NN / 32), 32, 0, stream>>>(dbl, LDBL, pk_wdt, DI, b_dt,
                                                          dtb, DI, 32, 2);
    // --- selective scan -> ybuf (includes +D*x and *silu(z)) ---
    mamba_scan<<<1, DI, 0, stream>>>(dtb, dbl, xbuf, xz, A_log, D_p, ybuf);

    // --- h_mamba = y @ W_out ---
    wmma_gemm8<<<dim3(HH / 64, NN / 32), 32, 0, stream>>>(ybuf, DI, pk_wout, HH, nullptr,
                                                          hmam, HH, DI, 0);
    // --- add + layernorm ---
    add_layernorm<<<(NN * 32 + 255) / 256, 256, 0, stream>>>(hgnn, hmam, gamma, beta, hfin);

    // --- graph mean pooling ---
    fill_kernel<<<(BB * HH + BB + 255) / 256, 256, 0, stream>>>(gesum, 0.0f, BB * HH + BB);
    seg_accum<<<(NN * HH + 255) / 256, 256, 0, stream>>>(hfin, batch, gesum, cnts);
    seg_div<<<(BB * HH + 255) / 256, 256, 0, stream>>>(gesum, cnts);

    // --- heads ---
    heads_kernel<<<(BB * TOTC + 255) / 256, 256, 0, stream>>>(gesum,
        (const float*)d_in[20], (const float*)d_in[21],
        (const float*)d_in[22], (const float*)d_in[23],
        (const float*)d_in[24], (const float*)d_in[25],
        (const float*)d_in[26], (const float*)d_in[27],
        (const float*)d_in[28], (const float*)d_in[29],
        (const float*)d_in[30], (const float*)d_in[31],
        (const float*)d_in[32], (const float*)d_in[33], out);
}
